// SphereAttention_75531294868082
// MI455X (gfx1250) — compile-verified, matrix operand order checked by reading, not run on hardware
//
#include <hip/hip_runtime.h>
#include <hip/hip_bf16.h>
#include <hip/hip_fp16.h>

// Problem constants
#define HH 128
#define WW 256
#define HWP (HH * WW)      // 32768 pixels
#define CC 256             // channels
#define NH 8               // heads
#define DH 32              // head dim
#define KT 9               // taps per pixel (3x3)
#define GXD (WW * 3)       // grid x extent = 768
#define GYD (HH * 3)       // grid y extent = 384

// Workspace layout (bytes)
#define XT_OFF   0u                         // xt: 32768*256 f16 = 16 MB
#define ATT_OFF  16777216u                  // att: 32768*256 f16 = 16 MB
#define WQ_OFF   33554432u                  // each weight: 256*256 f16 = 128 KB
#define WK_OFF   (WQ_OFF + 131072u)
#define WV_OFF   (WK_OFF + 131072u)
#define WO_OFF   (WV_OFF + 131072u)
#define IDX_OFF  (WO_OFF + 131072u)         // idx: 384*768 i32

typedef __attribute__((ext_vector_type(16))) _Float16 v16h;
typedef __attribute__((ext_vector_type(8)))  _Float16 v8h;
typedef __attribute__((ext_vector_type(8)))  float    v8f;

#define LDA 272   // padded LDS row stride in halfs (544B; 544 % 256 = 32 -> conflict-free b128)

__device__ __forceinline__ v8f wmma_f16(v16h a, v16h b, v8f c) {
  return __builtin_amdgcn_wmma_f32_16x16x32_f16(false, a, false, b, (short)0, c, false, false);
}

// A-fragment (16x32 f16) from LDS per documented layout:
// lane m=lane&15, half h=lane>>4; halfs [K=8h..8h+7] -> a[0..7], [K=16+8h..+7] -> a[8..15]
__device__ __forceinline__ v16h ld_afrag(const _Float16* row0, int kb, int lane) {
  int m = lane & 15, h = lane >> 4;
  const _Float16* p = row0 + m * LDA + kb * 32 + h * 8;
  v8h lo = *(const v8h*)p;
  v8h hi = *(const v8h*)(p + 16);
  v16h r;
#pragma unroll
  for (int i = 0; i < 8; ++i) { r[i] = lo[i]; r[8 + i] = hi[i]; }
  return r;
}

// B-fragment (32x16 f16) from a row-major [o][c] f16 weight matrix (mirrored layout, lane = n col)
__device__ __forceinline__ v16h ld_bfrag(const _Float16* __restrict__ w, int nbase, int kb, int lane) {
  int nn = lane & 15, h = lane >> 4;
  const _Float16* p = w + (size_t)(nbase + nn) * CC + kb * 32 + h * 8;
  v8h lo = *(const v8h*)p;
  v8h hi = *(const v8h*)(p + 16);
  v16h r;
#pragma unroll
  for (int i = 0; i < 8; ++i) { r[i] = lo[i]; r[8 + i] = hi[i]; }
  return r;
}

// ---------------- prep: gather indices from grid ----------------
__global__ void idx_kernel(const float* __restrict__ grid, int* __restrict__ idx2) {
  int g = blockIdx.x * 256 + threadIdx.x;
  if (g >= GYD * GXD) return;
  float gx = grid[2 * g + 0];
  float gy = grid[2 * g + 1];
  int ix = (int)rintf((gx + 1.0f) * 0.5f * (float)(WW - 1));
  int iy = (int)rintf((gy + 1.0f) * 0.5f * (float)(HH - 1));
  ix = min(max(ix, 0), WW - 1);
  iy = min(max(iy, 0), HH - 1);
  idx2[g] = iy * WW + ix;
}

// ---------------- prep: x (C,H,W) f32 -> xt (HW, C) f16, LDS-tiled transpose ----------------
__global__ __launch_bounds__(256) void xt_kernel(const float* __restrict__ x, _Float16* __restrict__ xt) {
  __shared__ float t[32][33];
  int p0 = blockIdx.x * 32;
  int c0 = blockIdx.y * 32;
  int a = threadIdx.x & 31, b = threadIdx.x >> 5; // b in 0..7
#pragma unroll
  for (int i = 0; i < 4; ++i)
    t[b + 8 * i][a] = x[(size_t)(c0 + b + 8 * i) * HWP + p0 + a];
  __syncthreads();
#pragma unroll
  for (int i = 0; i < 4; ++i)
    xt[(size_t)(p0 + b + 8 * i) * CC + c0 + a] = (_Float16)t[a][b + 8 * i];
}

// ---------------- prep: f32 -> f16 weight convert ----------------
__global__ void wconv_kernel(const float* __restrict__ w, _Float16* __restrict__ w16) {
  int i = blockIdx.x * 256 + threadIdx.x;
  if (i < CC * CC) w16[i] = (_Float16)w[i];
}

// ---------------- fused QKV GEMM + attention: one WG per 16-pixel tile, one wave per head ----------------
__global__ __launch_bounds__(256) void qkv_attn_kernel(
    const _Float16* __restrict__ xt, const int* __restrict__ idx2,
    const _Float16* __restrict__ wq16, const _Float16* __restrict__ wk16, const _Float16* __restrict__ wv16,
    const float* __restrict__ bq, const float* __restrict__ bk, const float* __restrict__ bv,
    _Float16* __restrict__ att)
{
  __shared__ __align__(16) _Float16 sA[144 * LDA];     // gathered samples, 78336 B
  __shared__ __align__(16) _Float16 sQx[16 * LDA];     // pixel rows, 8704 B
  __shared__ _Float16 sK[NH * 144 * 36];               // 82944 B
  __shared__ _Float16 sV[NH * 144 * 36];               // 82944 B
  __shared__ float    sQ[16 * 264];                    // 16896 B

  const int tid = threadIdx.x;
  const int lane = tid & 31;
  const int wave = tid >> 5;            // head id
  const int pbase = blockIdx.x * 16;

  // stage pixel rows (16 x 512B)
#pragma unroll
  for (int i = 0; i < 2; ++i) {
    int q = tid + i * 256;
    int r = q >> 5, ch = q & 31;
    *(v8h*)&sQx[r * LDA + ch * 8] = *(const v8h*)&xt[(size_t)(pbase + r) * CC + ch * 8];
  }
  // stage gathered sample rows (144 x 512B)
#pragma unroll
  for (int i = 0; i < 18; ++i) {
    int q = tid + i * 256;
    int s = q >> 5, ch = q & 31;
    int pl = s / 9, kk = s - pl * 9;
    int p = pbase + pl;
    int hy = p >> 8, wx = p & 255;      // W == 256
    int kh = kk / 3, kw = kk - kh * 3;
    int sp = idx2[(hy * 3 + kh) * GXD + (wx * 3 + kw)];
    *(v8h*)&sA[s * LDA + ch * 8] = *(const v8h*)&xt[(size_t)sp * CC + ch * 8];
  }
  __syncthreads();

  const int nn = lane & 15, hh = lane >> 4;
  const int n0 = wave * DH;
  const float bq0 = bq[n0 + nn], bq1 = bq[n0 + 16 + nn];
  const float bk0 = bk[n0 + nn], bk1 = bk[n0 + 16 + nn];
  const float bv0 = bv[n0 + nn], bv1 = bv[n0 + 16 + nn];

  v16h bf[2][8];

  // ---- Q = sQx * wq(head)^T + bq ----
#pragma unroll
  for (int nb = 0; nb < 2; ++nb)
#pragma unroll
    for (int kb = 0; kb < 8; ++kb)
      bf[nb][kb] = ld_bfrag(wq16, n0 + nb * 16, kb, lane);
  {
    v8f acc0 = {}, acc1 = {};
#pragma unroll
    for (int kb = 0; kb < 8; ++kb) {
      v16h a = ld_afrag(&sQx[0], kb, lane);
      acc0 = wmma_f16(a, bf[0][kb], acc0);
      acc1 = wmma_f16(a, bf[1][kb], acc1);
    }
#pragma unroll
    for (int g = 0; g < 8; ++g) {
      int pr = g + 8 * hh;
      sQ[pr * 264 + n0 + nn]      = acc0[g] + bq0;
      sQ[pr * 264 + n0 + 16 + nn] = acc1[g] + bq1;
    }
  }

  // ---- K = sA * wk(head)^T + bk ----
#pragma unroll
  for (int nb = 0; nb < 2; ++nb)
#pragma unroll
    for (int kb = 0; kb < 8; ++kb)
      bf[nb][kb] = ld_bfrag(wk16, n0 + nb * 16, kb, lane);
  for (int mb = 0; mb < 9; ++mb) {
    v8f acc0 = {}, acc1 = {};
#pragma unroll
    for (int kb = 0; kb < 8; ++kb) {
      v16h a = ld_afrag(&sA[mb * 16 * LDA], kb, lane);
      acc0 = wmma_f16(a, bf[0][kb], acc0);
      acc1 = wmma_f16(a, bf[1][kb], acc1);
    }
#pragma unroll
    for (int g = 0; g < 8; ++g) {
      int s = mb * 16 + g + 8 * hh;
      sK[(wave * 144 + s) * 36 + nn]      = (_Float16)(acc0[g] + bk0);
      sK[(wave * 144 + s) * 36 + 16 + nn] = (_Float16)(acc1[g] + bk1);
    }
  }

  // ---- V = sA * wv(head)^T + bv ----
#pragma unroll
  for (int nb = 0; nb < 2; ++nb)
#pragma unroll
    for (int kb = 0; kb < 8; ++kb)
      bf[nb][kb] = ld_bfrag(wv16, n0 + nb * 16, kb, lane);
  for (int mb = 0; mb < 9; ++mb) {
    v8f acc0 = {}, acc1 = {};
#pragma unroll
    for (int kb = 0; kb < 8; ++kb) {
      v16h a = ld_afrag(&sA[mb * 16 * LDA], kb, lane);
      acc0 = wmma_f16(a, bf[0][kb], acc0);
      acc1 = wmma_f16(a, bf[1][kb], acc1);
    }
#pragma unroll
    for (int g = 0; g < 8; ++g) {
      int s = mb * 16 + g + 8 * hh;
      sV[(wave * 144 + s) * 36 + nn]      = (_Float16)(acc0[g] + bv0);
      sV[(wave * 144 + s) * 36 + 16 + nn] = (_Float16)(acc1[g] + bv1);
    }
  }

  // ---- attention (per wave = head, lane = dim d) ----
  const float scale = 0.17677669529663687f; // 1/sqrt(32)
#pragma unroll 1
  for (int p = 0; p < 16; ++p) {
    float qv = sQ[p * 264 + n0 + lane];
    float lg[9];
    float mx = -1e30f;
#pragma unroll
    for (int k = 0; k < 9; ++k) {
      float pr = qv * (float)sK[(wave * 144 + p * 9 + k) * 36 + lane];
      pr += __shfl_xor(pr, 16, 32);
      pr += __shfl_xor(pr, 8, 32);
      pr += __shfl_xor(pr, 4, 32);
      pr += __shfl_xor(pr, 2, 32);
      pr += __shfl_xor(pr, 1, 32);
      lg[k] = pr * scale;
      mx = fmaxf(mx, lg[k]);
    }
    float se = 0.0f, ex[9];
#pragma unroll
    for (int k = 0; k < 9; ++k) { ex[k] = __expf(lg[k] - mx); se += ex[k]; }
    float inv = 1.0f / se;
    float o = 0.0f;
#pragma unroll
    for (int k = 0; k < 9; ++k)
      o += ex[k] * (float)sV[(wave * 144 + p * 9 + k) * 36 + lane];
    o *= inv;
    att[(size_t)(pbase + p) * CC + n0 + lane] = (_Float16)o;
  }
}

// ---------------- output GEMM: y[o][p] = sum_c att[p][c] * wo[o][c] + bo[o] ----------------
__global__ __launch_bounds__(256) void out_gemm_kernel(
    const _Float16* __restrict__ att, const _Float16* __restrict__ wo16,
    const float* __restrict__ bo, float* __restrict__ y)
{
  __shared__ __align__(16) _Float16 sA[128 * LDA];   // 69632 B
  const int tid = threadIdx.x, lane = tid & 31, wave = tid >> 5;
  const int pbase = blockIdx.x * 128;
#pragma unroll
  for (int i = 0; i < 16; ++i) {
    int q = tid + i * 256;
    int r = q >> 5, ch = q & 31;
    *(v8h*)&sA[r * LDA + ch * 8] = *(const v8h*)&att[(size_t)(pbase + r) * CC + ch * 8];
  }
  __syncthreads();

  const int nn = lane & 15, hh = lane >> 4;
  const int n0 = wave * 32;
  const float b0 = bo[n0 + nn], b1 = bo[n0 + 16 + nn];

  v16h bf[2][8];
#pragma unroll
  for (int nb = 0; nb < 2; ++nb)
#pragma unroll
    for (int kb = 0; kb < 8; ++kb)
      bf[nb][kb] = ld_bfrag(wo16, n0 + nb * 16, kb, lane);

  for (int mb = 0; mb < 8; ++mb) {
    v8f acc0 = {}, acc1 = {};
#pragma unroll
    for (int kb = 0; kb < 8; ++kb) {
      v16h a = ld_afrag(&sA[mb * 16 * LDA], kb, lane);
      acc0 = wmma_f16(a, bf[0][kb], acc0);
      acc1 = wmma_f16(a, bf[1][kb], acc1);
    }
    int p0 = pbase + mb * 16 + 8 * hh;   // consecutive g -> consecutive pixels
    {
      float4 lo; lo.x = acc0[0] + b0; lo.y = acc0[1] + b0; lo.z = acc0[2] + b0; lo.w = acc0[3] + b0;
      float4 hi; hi.x = acc0[4] + b0; hi.y = acc0[5] + b0; hi.z = acc0[6] + b0; hi.w = acc0[7] + b0;
      float* dst = y + (size_t)(n0 + nn) * HWP + p0;
      *(float4*)dst = lo;
      *(float4*)(dst + 4) = hi;
    }
    {
      float4 lo; lo.x = acc1[0] + b1; lo.y = acc1[1] + b1; lo.z = acc1[2] + b1; lo.w = acc1[3] + b1;
      float4 hi; hi.x = acc1[4] + b1; hi.y = acc1[5] + b1; hi.z = acc1[6] + b1; hi.w = acc1[7] + b1;
      float* dst = y + (size_t)(n0 + 16 + nn) * HWP + p0;
      *(float4*)dst = lo;
      *(float4*)(dst + 4) = hi;
    }
  }
}

extern "C" void kernel_launch(void* const* d_in, const int* in_sizes, int n_in,
                              void* d_out, int out_size, void* d_ws, size_t ws_size,
                              hipStream_t stream) {
  (void)in_sizes; (void)n_in; (void)out_size; (void)ws_size;
  const float* x    = (const float*)d_in[0];
  const float* grid = (const float*)d_in[1];
  const float* wq   = (const float*)d_in[2];
  const float* bq   = (const float*)d_in[3];
  const float* wk   = (const float*)d_in[4];
  const float* bk   = (const float*)d_in[5];
  const float* wv   = (const float*)d_in[6];
  const float* bv   = (const float*)d_in[7];
  const float* wo   = (const float*)d_in[8];
  const float* bo   = (const float*)d_in[9];

  char* ws = (char*)d_ws;
  _Float16* xt   = (_Float16*)(ws + XT_OFF);
  _Float16* att  = (_Float16*)(ws + ATT_OFF);
  _Float16* wq16 = (_Float16*)(ws + WQ_OFF);
  _Float16* wk16 = (_Float16*)(ws + WK_OFF);
  _Float16* wv16 = (_Float16*)(ws + WV_OFF);
  _Float16* wo16 = (_Float16*)(ws + WO_OFF);
  int*      idx2 = (int*)(ws + IDX_OFF);
  float*    y    = (float*)d_out;

  idx_kernel<<<(GYD * GXD + 255) / 256, 256, 0, stream>>>(grid, idx2);
  dim3 tg(HWP / 32, CC / 32);
  xt_kernel<<<tg, 256, 0, stream>>>(x, xt);
  wconv_kernel<<<256, 256, 0, stream>>>(wq, wq16);
  wconv_kernel<<<256, 256, 0, stream>>>(wk, wk16);
  wconv_kernel<<<256, 256, 0, stream>>>(wv, wv16);
  wconv_kernel<<<256, 256, 0, stream>>>(wo, wo16);

  qkv_attn_kernel<<<HWP / 16, 256, 0, stream>>>(xt, idx2, wq16, wk16, wv16, bq, bk, bv, att);
  out_gemm_kernel<<<HWP / 128, 256, 0, stream>>>(att, wo16, bo, y);
}
